// Encoder_73263552135578
// MI455X (gfx1250) — matrix operand order repaired and checked by reading, not compiled, not run on hardware
//
#include <hip/hip_runtime.h>
#include <hip/hip_bf16.h>

// ---------------------------------------------------------------------------
// Problem constants (from reference)
// ---------------------------------------------------------------------------
constexpr int   cB    = 4;
constexpr int   cL    = 512;
constexpr int   cHID  = 512;
constexpr int   cNH   = 8;
constexpr int   cFF   = 2048;
constexpr int   cNL   = 4;
constexpr int   cKQ   = 8;
constexpr int   cKV   = 12;
constexpr float cNORM = 3000.0f;
constexpr int   cTOK  = cB * cL;        // 2048 tokens
constexpr int   cDTOK = cTOK * 20;      // 40960 "dist" rows

__device__ __constant__ float c_AA[20] = {
  71.03711f, 156.10111f, 114.04293f, 115.02694f, 103.00919f, 129.04259f,
  128.05858f, 57.02146f, 137.05891f, 113.08406f, 113.08406f, 128.09496f,
  131.04049f, 147.06841f, 97.05276f, 87.03203f, 101.04768f, 99.06841f,
  186.07931f, 163.06333f };

// ---------------------------------------------------------------------------
// WMMA GEMM: C[M,N] = act(A[M,Kp] @ Wt[N,Kp]^T + bias)
//  - A fp32 row-major, padded lda (%32==0, pad zeroed) -> reg path (needs cvt)
//  - Wt bf16 N-major pre-transposed, ldb (%32==0, pad zeroed)
//      -> staged with GLOBAL_LOAD_ASYNC_TO_LDS_B128 (ASYNCcnt), no VGPRs
//  - M % 128 == 0, N % 128 == 0 guaranteed by caller -> no guards
//  - block tile 128x128, 8 waves (2x4), wave tile 64x32 = 8 WMMA/K-step
//  - LDS double-buffered, one barrier per K-step
// ---------------------------------------------------------------------------
typedef __attribute__((ext_vector_type(16))) __bf16 v16bf;
typedef __attribute__((ext_vector_type(8)))  float  v8f;
union Frag16 { v16bf v; uint4 u[2]; };
union Pack16 { __bf16 h[16]; uint4 u[2]; };

constexpr int BM = 128, BN_ = 128, BK = 32, LDSA = 40;  // padded LDS row stride

__device__ __forceinline__ void async_copy_b128x2(unsigned lds_addr, const void* gaddr) {
  // INST_OFFSET is added to both the LDS destination and the global address,
  // so offset:16 moves the second 16-byte half of this thread's 32-byte slot.
  asm volatile(
      "global_load_async_to_lds_b128 %0, %1, off\n"
      "global_load_async_to_lds_b128 %0, %1, off offset:16"
      :: "v"(lds_addr), "v"((unsigned long long)gaddr) : "memory");
}
__device__ __forceinline__ void wait_async0() {
  asm volatile("s_wait_asynccnt 0" ::: "memory");
}

__global__ __launch_bounds__(256) void gemm_bf16_wmma(
    const float* __restrict__ A, int lda,
    const __bf16* __restrict__ Wt, int ldb,
    float* __restrict__ C, const float* __restrict__ bias,
    int M, int N, int act)
{
  __shared__ __align__(16) __bf16 As[2][BM  * LDSA];
  __shared__ __align__(16) __bf16 Bs[2][BN_ * LDSA];
  const int tid  = threadIdx.x;
  const int lane = tid & 31, wave = tid >> 5;
  const int bm = blockIdx.x * BM, bn = blockIdx.y * BN_;
  const int wm = (wave >> 2) * 64;     // 0 / 64
  const int wn = (wave & 3) * 32;      // 0 / 32 / 64 / 96
  const int lrow = lane & 15, lhi = lane >> 4;

  // staging: both tiles are 128 rows x 32 cols; 2 threads per row
  const int srow = tid >> 1;
  const int skf  = (tid & 1) * 16;
  const float*  aptr = A  + (size_t)(bm + srow) * lda + skf;
  const __bf16* bptr = Wt + (size_t)(bn + srow) * ldb + skf;
  const unsigned bs_lds[2] = {
    (unsigned)(unsigned long long)&Bs[0][srow * LDSA + skf],
    (unsigned)(unsigned long long)&Bs[1][srow * LDSA + skf] };

  const int T = ldb / BK;   // ldb == padded K

  v8f acc[4][2] = {};

  // ---- prologue: stage tile 0 (B async, A through regs w/ convert) ----
  {
    async_copy_b128x2(bs_lds[0], bptr);
    float4 f0 = *(const float4*)(aptr + 0);
    float4 f1 = *(const float4*)(aptr + 4);
    float4 f2 = *(const float4*)(aptr + 8);
    float4 f3 = *(const float4*)(aptr + 12);
    Pack16 pk;
    pk.h[0]=(__bf16)f0.x; pk.h[1]=(__bf16)f0.y; pk.h[2]=(__bf16)f0.z; pk.h[3]=(__bf16)f0.w;
    pk.h[4]=(__bf16)f1.x; pk.h[5]=(__bf16)f1.y; pk.h[6]=(__bf16)f1.z; pk.h[7]=(__bf16)f1.w;
    pk.h[8]=(__bf16)f2.x; pk.h[9]=(__bf16)f2.y; pk.h[10]=(__bf16)f2.z; pk.h[11]=(__bf16)f2.w;
    pk.h[12]=(__bf16)f3.x; pk.h[13]=(__bf16)f3.y; pk.h[14]=(__bf16)f3.z; pk.h[15]=(__bf16)f3.w;
    uint4* da = (uint4*)&As[0][srow * LDSA + skf];
    da[0] = pk.u[0]; da[1] = pk.u[1];
    wait_async0();
  }
  __syncthreads();

  for (int t = 0; t < T; ++t) {
    const int cur = t & 1, nxt = cur ^ 1;
    const bool more = (t + 1) < T;

    // fire next-tile loads first so they overlap the WMMAs below
    float4 f0, f1, f2, f3;
    if (more) {
      async_copy_b128x2(bs_lds[nxt], bptr + (t + 1) * BK);   // B: DMA to LDS
      const float* ap = aptr + (t + 1) * BK;                 // A: regs (cvt)
      f0 = *(const float4*)(ap + 0);
      f1 = *(const float4*)(ap + 4);
      f2 = *(const float4*)(ap + 8);
      f3 = *(const float4*)(ap + 12);
    }

    // fragments (documented CDNA5 16-bit layouts):
    // A 16x32: lane row = lane%16, K-octets (lane/16)*8 and +16
    // B 32x16: lane col = lane%16, K-octets (lane/16)*8 and +16
    v16bf afrag[4], bfrag[2];
#pragma unroll
    for (int i = 0; i < 4; ++i) {
      Frag16 f;
      const int r = wm + i * 16 + lrow;
      f.u[0] = *(const uint4*)&As[cur][r * LDSA + lhi * 8];
      f.u[1] = *(const uint4*)&As[cur][r * LDSA + lhi * 8 + 16];
      afrag[i] = f.v;
    }
#pragma unroll
    for (int j = 0; j < 2; ++j) {
      Frag16 f;
      const int n = wn + j * 16 + lrow;
      f.u[0] = *(const uint4*)&Bs[cur][n * LDSA + lhi * 8];
      f.u[1] = *(const uint4*)&Bs[cur][n * LDSA + lhi * 8 + 16];
      bfrag[j] = f.v;
    }
#pragma unroll
    for (int i = 0; i < 4; ++i)
#pragma unroll
      for (int j = 0; j < 2; ++j)
        acc[i][j] = __builtin_amdgcn_wmma_f32_16x16x32_bf16(
            false, afrag[i], false, bfrag[j], (short)0, acc[i][j], false, false);

    if (more) {
      Pack16 pk;
      pk.h[0]=(__bf16)f0.x; pk.h[1]=(__bf16)f0.y; pk.h[2]=(__bf16)f0.z; pk.h[3]=(__bf16)f0.w;
      pk.h[4]=(__bf16)f1.x; pk.h[5]=(__bf16)f1.y; pk.h[6]=(__bf16)f1.z; pk.h[7]=(__bf16)f1.w;
      pk.h[8]=(__bf16)f2.x; pk.h[9]=(__bf16)f2.y; pk.h[10]=(__bf16)f2.z; pk.h[11]=(__bf16)f2.w;
      pk.h[12]=(__bf16)f3.x; pk.h[13]=(__bf16)f3.y; pk.h[14]=(__bf16)f3.z; pk.h[15]=(__bf16)f3.w;
      uint4* da = (uint4*)&As[nxt][srow * LDSA + skf];
      da[0] = pk.u[0]; da[1] = pk.u[1];
      wait_async0();   // our async B copies for buffer nxt have landed
    }
    __syncthreads();
  }

  // Epilogue. C/D layout: vgpr r, lane -> (M = r + 8*(lane/16), N = lane%16)
#pragma unroll
  for (int j = 0; j < 2; ++j) {
    const int n = bn + wn + j * 16 + lrow;
    const float bv = bias ? bias[n] : 0.0f;
#pragma unroll
    for (int i = 0; i < 4; ++i) {
#pragma unroll
      for (int r = 0; r < 8; ++r) {
        const int m = bm + wm + i * 16 + lhi * 8 + r;
        float v = acc[i][j][r] + bv;
        if (act == 1)      v = fmaxf(v, 0.0f);
        else if (act == 2) v = 0.5f * v * (1.0f + tanhf(0.7978845608f * (v + 0.044715f * v * v * v)));
        C[(size_t)m * N + n] = v;
      }
    }
  }
}

// Weight pre-pass: W[K,N] fp32 -> Wt[N,Kp] bf16 (transposed, K padded to 32)
__global__ void convert_wt_kernel(const float* __restrict__ W, __bf16* __restrict__ Wt,
                                  int K, int N, int Kp) {
  size_t t = (size_t)blockIdx.x * blockDim.x + threadIdx.x;
  if (t >= (size_t)N * Kp) return;
  int n = (int)(t / Kp), k = (int)(t % Kp);
  Wt[t] = (k < K) ? (__bf16)W[(size_t)k * N + n] : (__bf16)0.0f;
}

// ---------------------------------------------------------------------------
// Embedding helpers
// ---------------------------------------------------------------------------
__device__ __forceinline__ int write_embed(float* o, int p, float x) {
  o[p++] = x;
  float f = 1.0f;
#pragma unroll
  for (int k = 0; k < 10; ++k) { float xs = x * f; o[p++] = __sinf(xs); o[p++] = __cosf(xs); f *= 2.0f; }
  const float xn = x * cNORM;
  f = 1.0f;
#pragma unroll
  for (int k = 0; k < 10; ++k) { float xs = xn * f; o[p++] = __sinf(xs); o[p++] = __cosf(xs); f *= 2.0f; }
  return p;  // +41
}

// x0 row stride 96 (83 used, pad zeroed) so the GEMM needs no K guards
__global__ void build_x0_kernel(const float* __restrict__ mz, const float* __restrict__ inten,
                                const float* __restrict__ h_mass, float* __restrict__ mzn,
                                float* __restrict__ x0) {
  int t = blockIdx.x * blockDim.x + threadIdx.x;
  if (t >= cTOK) return;
  int b = t / cL;
  float m = mz[t] / cNORM;
  mzn[t] = m;
  float tot = 0.0f;
  for (int j = 0; j < 8; ++j) tot += h_mass[b * 8 + j];
  tot /= cNORM;
  float* o = x0 + (size_t)t * 96;
  int p = write_embed(o, 0, m);
  o[p++] = inten[t];
  p = write_embed(o, p, tot);
  for (; p < 96; ++p) o[p] = 0.0f;
}

// ---------------------------------------------------------------------------
// KNN over spectrum (KQ=8) — one thread per query token
// ---------------------------------------------------------------------------
__global__ void knn_q_kernel(const float* __restrict__ mzn, const int* __restrict__ p_len,
                             int* __restrict__ knn_idx, float* __restrict__ kd) {
  int t = blockIdx.x * blockDim.x + threadIdx.x;
  if (t >= cTOK) return;
  int b = t / cL, i = t % cL;
  int len = p_len[b];
  float qi = mzn[t];
  float bv[cKQ]; int bix[cKQ];
#pragma unroll
  for (int s = 0; s < cKQ; ++s) { bv[s] = -1e30f; bix[s] = 0; }
  for (int j = 0; j < len; ++j) {
    float d = qi - mzn[b * cL + j];
    float neg = -d * d;
    if (neg > bv[cKQ - 1]) {
      int s = cKQ - 1;
      while (s > 0 && neg > bv[s - 1]) { bv[s] = bv[s - 1]; bix[s] = bix[s - 1]; --s; }
      bv[s] = neg; bix[s] = j;
    }
  }
  bool mq = (i >= len);
  for (int s = 0; s < cKQ; ++s) {
    int idx = mq ? 0 : bix[s];
    float d2 = fmaxf(-bv[s], 0.0f);
    float diff = mzn[b * cL + idx] - qi;
    float sg = (diff > 0.f) ? 1.f : (diff < 0.f ? -1.f : 0.f);
    kd[t * cKQ + s] = mq ? 0.f : sqrtf(d2) * sg;
    knn_idx[t * cKQ + s] = idx;
  }
}

// ---------------------------------------------------------------------------
// get_aa: 80 ion queries x KNN-3, then top-KV of 240  — one block per token
// ---------------------------------------------------------------------------
__global__ __launch_bounds__(128) void get_aa_kernel(
    const float* __restrict__ mzn, const int* __restrict__ p_len,
    int* __restrict__ aa_idx, float* __restrict__ aa_val,
    int* __restrict__ aa_col, int* __restrict__ maskq) {
  const int t = blockIdx.x;
  const int b = t / cL, i = t % cL;
  __shared__ float s_ref[cL];
  __shared__ float s_dneg[240];
  __shared__ float s_sgn[240];
  __shared__ int   s_ri[240];
  for (int j = threadIdx.x; j < cL; j += 128) s_ref[j] = mzn[b * cL + j];
  __syncthreads();
  const int len = p_len[b];
  const float qi = mzn[t];
  for (int a = threadIdx.x; a < 80; a += 128) {
    float aa = c_AA[a % 20];
    int g = a / 20;
    float sc = (g == 0) ? 1.f : (g == 1) ? -1.f : (g == 2) ? 0.5f : -0.5f;
    float ion = qi + aa * sc / cNORM;
    bool bad = (ion > 1.f) || (ion < 0.f);
    float b0 = -1e30f, b1 = -1e30f, b2 = -1e30f; int i0 = 0, i1 = 0, i2 = 0;
    for (int j = 0; j < len; ++j) {
      float d = ion - s_ref[j];
      float neg = -d * d;
      if (neg > b0)      { b2 = b1; i2 = i1; b1 = b0; i1 = i0; b0 = neg; i0 = j; }
      else if (neg > b1) { b2 = b1; i2 = i1; b1 = neg; i1 = j; }
      else if (neg > b2) { b2 = neg; i2 = j; }
    }
    float nn[3] = { b0, b1, b2 }; int ii[3] = { i0, i1, i2 };
#pragma unroll
    for (int kk = 0; kk < 3; ++kk) {
      float d2 = fmaxf(-nn[kk], 0.f);
      float diff = s_ref[ii[kk]] - ion;
      float sg = (diff > 0.f) ? 1.f : (diff < 0.f ? -1.f : 0.f);
      s_dneg[a * 3 + kk] = bad ? -10.f : -sqrtf(d2);
      s_sgn[a * 3 + kk]  = sg;
      s_ri[a * 3 + kk]   = ii[kk];
    }
  }
  __syncthreads();
  if (threadIdx.x == 0) {
    float bv[cKV]; int bj[cKV];
#pragma unroll
    for (int s = 0; s < cKV; ++s) { bv[s] = -1e30f; bj[s] = 0; }
    for (int c = 0; c < 240; ++c) {
      float v = s_dneg[c];
      if (v > bv[cKV - 1]) {
        int s = cKV - 1;
        while (s > 0 && v > bv[s - 1]) { bv[s] = bv[s - 1]; bj[s] = bj[s - 1]; --s; }
        bv[s] = v; bj[s] = c;
      }
    }
    const bool mq = (i >= len);
    maskq[t] = mq ? 1 : 0;
    for (int s = 0; s < cKV; ++s) {
      int c = bj[s];
      aa_val[t * cKV + s] = mq ? 0.f : fabsf(s_dneg[c]) * s_sgn[c];
      aa_col[t * cKV + s] = mq ? -1 : (c / 3);
      aa_idx[t * cKV + s] = mq ? 0 : s_ri[c];
    }
  }
}

// din row stride 96 (66 used, pad zeroed): [embed(val) (41), val, extras 0..23]
__global__ void build_din_kernel(const float* __restrict__ kd, const float* __restrict__ aa_val,
                                 const int* __restrict__ aa_col, float* __restrict__ din) {
  int t = blockIdx.x * blockDim.x + threadIdx.x;
  if (t >= cDTOK) return;
  int s = t % 20, bi = t / 20;
  float val; int col;
  if (s < cKQ) { val = kd[bi * cKQ + s]; col = -1; }
  else         { val = aa_val[bi * cKV + (s - cKQ)]; col = aa_col[bi * cKV + (s - cKQ)]; }
  float* o = din + (size_t)t * 96;
  int p = write_embed(o, 0, val);
  o[p] = val;
#pragma unroll
  for (int c = 0; c < 24; ++c) o[42 + c] = 0.f;
  if (col >= 0) { o[42 + (col % 20)] = 1.f; o[42 + 20 + (col / 20)] = 1.f; }
  for (int c = 66; c < 96; ++c) o[c] = 0.f;
}

__global__ void mask_dist_kernel(float* __restrict__ dist, const int* __restrict__ maskq) {
  size_t t = (size_t)blockIdx.x * blockDim.x + threadIdx.x;
  if (t >= (size_t)cDTOK * cHID) return;
  int bi = (int)(t / (20 * cHID));
  if (maskq[bi]) dist[t] = 0.f;
}

// ---------------------------------------------------------------------------
// LayerNorm — one block (8 waves) per token
// ---------------------------------------------------------------------------
__global__ __launch_bounds__(256) void layernorm_kernel(
    const float* __restrict__ x, const float* __restrict__ g,
    const float* __restrict__ b, float* __restrict__ y, int ncol) {
  const int t = blockIdx.x;
  const float* xi = x + (size_t)t * ncol;
  float* yi = y + (size_t)t * ncol;
  __shared__ float sred[8];
  __shared__ float s_mean, s_rstd;
  const int lane = threadIdx.x & 31, wave = threadIdx.x >> 5;
  float s = 0.f;
  for (int d = threadIdx.x; d < ncol; d += 256) s += xi[d];
  for (int o = 16; o > 0; o >>= 1) s += __shfl_xor(s, o, 32);
  if (lane == 0) sred[wave] = s;
  __syncthreads();
  if (threadIdx.x == 0) {
    float m = 0.f;
    for (int w = 0; w < 8; ++w) m += sred[w];
    s_mean = m / ncol;
  }
  __syncthreads();
  const float mean = s_mean;
  float v = 0.f;
  for (int d = threadIdx.x; d < ncol; d += 256) { float dd = xi[d] - mean; v += dd * dd; }
  for (int o = 16; o > 0; o >>= 1) v += __shfl_xor(v, o, 32);
  if (lane == 0) sred[wave] = v;
  __syncthreads();
  if (threadIdx.x == 0) {
    float m = 0.f;
    for (int w = 0; w < 8; ++w) m += sred[w];
    s_rstd = rsqrtf(m / ncol + 1e-5f);
  }
  __syncthreads();
  const float rstd = s_rstd;
  for (int d = threadIdx.x; d < ncol; d += 256) yi[d] = (xi[d] - mean) * rstd * g[d] + b[d];
}

// ---------------------------------------------------------------------------
// vmax: gated max-pool over KV gathered values — one thread per (token, d)
// pm rows are (bi*20+s)*1024;  dist_k = cols [0,512), dist_v = cols [512,1024)
// ---------------------------------------------------------------------------
__global__ void vmax_kernel(const float* __restrict__ qkvg, const float* __restrict__ pm,
                            const int* __restrict__ aa_idx, const int* __restrict__ maskq,
                            float* __restrict__ vmaxo) {
  int t = blockIdx.x * blockDim.x + threadIdx.x;
  if (t >= cTOK * cHID) return;
  int d = t % cHID, bi = t / cHID, b = bi / cL;
  bool mq = maskq[bi] != 0;
  float best = -1e30f;
#pragma unroll
  for (int jj = 0; jj < cKV; ++jj) {
    int ai = aa_idx[bi * cKV + jj];
    const size_t row = (size_t)(b * cL + ai) * (4 * cHID);
    float vv = mq ? 0.f : qkvg[row + 2 * cHID + d];
    float gg = mq ? 0.f : qkvg[row + 3 * cHID + d];
    const float* pmrow = pm + ((size_t)bi * 20 + cKQ + jj) * (2 * cHID);
    float gate = 1.f / (1.f + __expf(-(gg + pmrow[d])));
    best = fmaxf(best, gate * (vv + pmrow[cHID + d]));
  }
  vmaxo[(size_t)bi * cHID + d] = best;
}

// ---------------------------------------------------------------------------
// Attention over KQ=8 neighbors — one thread per (token, head)
// ---------------------------------------------------------------------------
__global__ void attn_kernel(const float* __restrict__ qkvg, const float* __restrict__ pm,
                            const float* __restrict__ vmaxi, const int* __restrict__ knn_idx,
                            float* __restrict__ agg) {
  int t = blockIdx.x * blockDim.x + threadIdx.x;
  if (t >= cTOK * cNH) return;
  constexpr int dk = cHID / cNH;  // 64
  int h = t % cNH, bi = t / cNH, b = bi / cL;
  const float* q = qkvg + (size_t)bi * (4 * cHID) + h * dk;
  float logits[cKQ];
  float mx = -1e30f;
  for (int j = 0; j < cKQ; ++j) {
    int kidx = knn_idx[bi * cKQ + j];
    const float* kk  = qkvg + (size_t)(b * cL + kidx) * (4 * cHID) + cHID + h * dk;
    const float* pqk = pm + ((size_t)bi * 20 + j) * (2 * cHID) + h * dk;
    float dot = 0.f;
    for (int d = 0; d < dk; ++d) dot += q[d] * (kk[d] + pqk[d]);
    logits[j] = dot * 0.125f;  // 1/sqrt(64)
    mx = fmaxf(mx, logits[j]);
  }
  float sum = 0.f;
  for (int j = 0; j < cKQ; ++j) { logits[j] = __expf(logits[j] - mx); sum += logits[j]; }
  float inv = 1.f / sum;
  float out[dk];
  for (int d = 0; d < dk; ++d) out[d] = 0.f;
  for (int j = 0; j < cKQ; ++j) {
    float w = logits[j] * inv;
    int kidx = knn_idx[bi * cKQ + j];
    const float* vv  = vmaxi + (size_t)(b * cL + kidx) * cHID + h * dk;
    const float* pqv = pm + ((size_t)bi * 20 + j) * (2 * cHID) + cHID + h * dk;
    for (int d = 0; d < dk; ++d) out[d] += w * (vv[d] + pqv[d]);
  }
  float* o = agg + (size_t)bi * cHID + h * dk;
  for (int d = 0; d < dk; ++d) o[d] = out[d];
}

// ---------------------------------------------------------------------------
// Elementwise helpers
// ---------------------------------------------------------------------------
__global__ void ff_gate_kernel(const float* __restrict__ h1, float* __restrict__ hg) {
  size_t t = (size_t)blockIdx.x * blockDim.x + threadIdx.x;
  if (t >= (size_t)cTOK * cFF) return;
  size_t tok = t / cFF, d = t % cFF;
  float a = h1[tok * (2 * cFF) + d];
  float b = h1[tok * (2 * cFF) + cFF + d];
  float g = 0.5f * a * (1.0f + tanhf(0.7978845608f * (a + 0.044715f * a * a * a)));
  hg[t] = g * b;
}

__global__ void add_inplace_kernel(float* __restrict__ x, const float* __restrict__ y, int n) {
  int t = blockIdx.x * blockDim.x + threadIdx.x;
  if (t < n) x[t] += y[t];
}

__global__ void final_out_kernel(const float* __restrict__ x, const float* __restrict__ ffo,
                                 const int* __restrict__ maskq, float* __restrict__ out) {
  int t = blockIdx.x * blockDim.x + threadIdx.x;
  if (t < cTOK * cHID) out[t] = 2.f * x[t] + ffo[t];        // x + pwff(...) + x
  if (t < cTOK) out[cTOK * cHID + t] = maskq[t] ? 1.f : 0.f; // maskq (b,l,1,1)
}

// ---------------------------------------------------------------------------
// Host orchestration
// ---------------------------------------------------------------------------
struct WtDesc { const __bf16* ptr; int ldb; };

extern "C" void kernel_launch(void* const* d_in, const int* in_sizes, int n_in,
                              void* d_out, int out_size, void* d_ws, size_t ws_size,
                              hipStream_t stream) {
  (void)in_sizes; (void)n_in; (void)out_size; (void)ws_size;
  auto F = [&](int i) { return (const float*)d_in[i]; };
  const float* mz     = F(0);
  const float* inten  = F(1);
  const float* h_mass = F(2);
  const int*   p_len  = (const int*)d_in[3];
  // params flattened depth-first in setup_inputs() insertion order
  enum { P_PRET_W1 = 4, P_PRET_W2, P_PRET_B2, P_PM_W, P_PM_B,
         P_PMFF_W1, P_PMFF_B1, P_PMFF_W2, P_PMFF_B2, P_LAYER0 = 13 };
  const int LSTRIDE = 14;                  // per-layer param count
  const int PF = P_LAYER0 + cNL * LSTRIDE; // fin_g, fin_b, fin_ff.{w1,b1,w2,b2}

  float* ws = (float*)d_ws;
  size_t off = 0;
  auto alloc = [&](size_t nfloats) { float* p = ws + off; off += nfloats; return p; };
  float* mzn    = alloc(cTOK);
  float* x0     = alloc((size_t)cTOK * 96);
  float* x      = alloc((size_t)cTOK * cHID);
  float* xn     = alloc((size_t)cTOK * cHID);
  float* qkvg   = alloc((size_t)cTOK * 4 * cHID);
  float* agg    = alloc((size_t)cTOK * cHID);
  float* vmaxb  = alloc((size_t)cTOK * cHID);
  float* h1     = alloc((size_t)cTOK * 2 * cFF);
  float* hg     = alloc((size_t)cTOK * cFF);
  float* din    = alloc((size_t)cDTOK * 96);
  float* dd1    = alloc((size_t)cDTOK * cHID);
  float* dd2    = alloc((size_t)cDTOK * 2048);   // gelu hidden; later pm1+pm
  float* dist   = alloc((size_t)cDTOK * cHID);
  float* kd     = alloc((size_t)cTOK * cKQ);
  float* aa_val = alloc((size_t)cTOK * cKV);
  int* knn_idx  = (int*)alloc(cTOK * cKQ);
  int* aa_idx   = (int*)alloc(cTOK * cKV);
  int* aa_col   = (int*)alloc(cTOK * cKV);
  int* maskq    = (int*)alloc(cTOK);
  float* pm1 = dd2;                              // 40960 x 1024
  float* pm  = dd2 + (size_t)cDTOK * (2 * cHID); // 40960 x 1024

  const int TPB = 256;
  auto blk = [&](long long n) { return (int)((n + TPB - 1) / TPB); };

  // bf16 transposed weights (converted once per launch, reused by all blocks)
  auto make_wt = [&](const float* W, int K, int N) {
    int Kp = (K + 31) & ~31;
    __bf16* p = (__bf16*)alloc(((size_t)N * Kp + 1) / 2);
    convert_wt_kernel<<<blk((long long)N * Kp), TPB, 0, stream>>>(W, p, K, N, Kp);
    return WtDesc{p, Kp};
  };
  WtDesc w_pret1 = make_wt(F(P_PRET_W1), 83, cFF);
  WtDesc w_pret2 = make_wt(F(P_PRET_W2), cFF, cHID);
  WtDesc w_pm    = make_wt(F(P_PM_W),    66, cHID);
  WtDesc w_pmff1 = make_wt(F(P_PMFF_W1), cHID, cFF);
  WtDesc w_pmff2 = make_wt(F(P_PMFF_W2), cFF, cHID);
  WtDesc w_qkvg[cNL], w_pos1[cNL], w_pos2[cNL], w_out[cNL], w_ff1[cNL], w_ff2[cNL];
  for (int l = 0; l < cNL; ++l) {
    const int LB = P_LAYER0 + l * LSTRIDE;
    w_qkvg[l] = make_wt(F(LB + 2), cHID, 4 * cHID);
    w_pos1[l] = make_wt(F(LB + 3), cHID, 2 * cHID);
    w_pos2[l] = make_wt(F(LB + 5), 2 * cHID, 2 * cHID);
    w_out[l]  = make_wt(F(LB + 7), cHID, cHID);
    w_ff1[l]  = make_wt(F(LB + 10), cHID, 2 * cFF);
    w_ff2[l]  = make_wt(F(LB + 12), cFF, cHID);
  }
  WtDesc w_fin1 = make_wt(F(PF + 2), cHID, cFF);
  WtDesc w_fin2 = make_wt(F(PF + 4), cFF, cHID);

  auto gemm = [&](const float* A, int lda, const WtDesc& w, float* C, const float* bias,
                  int M, int N, int act) {
    dim3 grid(M / BM, N / BN_);
    gemm_bf16_wmma<<<grid, 256, 0, stream>>>(A, lda, w.ptr, w.ldb, C, bias, M, N, act);
  };

  // ---- features / knn ----
  build_x0_kernel<<<blk(cTOK), TPB, 0, stream>>>(mz, inten, h_mass, mzn, x0);
  knn_q_kernel<<<blk(cTOK), TPB, 0, stream>>>(mzn, p_len, knn_idx, kd);
  get_aa_kernel<<<cTOK, 128, 0, stream>>>(mzn, p_len, aa_idx, aa_val, aa_col, maskq);
  build_din_kernel<<<blk(cDTOK), TPB, 0, stream>>>(kd, aa_val, aa_col, din);

  // ---- dist = pwff(relu(din @ pm_w + pm_b)), then mask ----
  gemm(din, 96, w_pm,    dd1,  F(P_PM_B),    cDTOK, cHID, 1);
  gemm(dd1, cHID, w_pmff1, dd2,  F(P_PMFF_B1), cDTOK, cFF,  2);
  gemm(dd2, cFF, w_pmff2, dist, F(P_PMFF_B2), cDTOK, cHID, 0);
  mask_dist_kernel<<<blk((long long)cDTOK * cHID), TPB, 0, stream>>>(dist, maskq);

  // ---- pre-transform: x = relu(relu(x0@w1)@w2 + b2) ----
  gemm(x0, 96, w_pret1, h1, nullptr,      cTOK, cFF,  1);
  gemm(h1, cFF, w_pret2, x,  F(P_PRET_B2), cTOK, cHID, 1);

  // ---- transformer layers ----
  for (int l = 0; l < cNL; ++l) {
    const int LB = P_LAYER0 + l * LSTRIDE;
    layernorm_kernel<<<cTOK, 256, 0, stream>>>(x, F(LB + 0), F(LB + 1), xn, cHID);
    gemm(xn, cHID, w_qkvg[l], qkvg, nullptr, cTOK, 4 * cHID, 0);
    // pos MLP over dist rows (dominant GEMMs)
    gemm(dist, cHID, w_pos1[l], pm1, F(LB + 4), cDTOK, 2 * cHID, 1);
    gemm(pm1, 2 * cHID, w_pos2[l], pm, F(LB + 6), cDTOK, 2 * cHID, 0);
    vmax_kernel<<<blk((long long)cTOK * cHID), TPB, 0, stream>>>(qkvg, pm, aa_idx, maskq, vmaxb);
    attn_kernel<<<blk((long long)cTOK * cNH), TPB, 0, stream>>>(qkvg, pm, vmaxb, knn_idx, agg);
    gemm(agg, cHID, w_out[l], xn, nullptr, cTOK, cHID, 0);
    add_inplace_kernel<<<blk(cTOK * cHID), TPB, 0, stream>>>(x, xn, cTOK * cHID);
    layernorm_kernel<<<cTOK, 256, 0, stream>>>(x, F(LB + 8), F(LB + 9), xn, cHID);
    gemm(xn, cHID, w_ff1[l], h1, F(LB + 11), cTOK, 2 * cFF, 0);
    ff_gate_kernel<<<blk((long long)cTOK * cFF), TPB, 0, stream>>>(h1, hg);
    gemm(hg, cFF, w_ff2[l], xn, F(LB + 13), cTOK, cHID, 0);
    add_inplace_kernel<<<blk(cTOK * cHID), TPB, 0, stream>>>(x, xn, cTOK * cHID);
  }

  // ---- final FF + output ----
  layernorm_kernel<<<cTOK, 256, 0, stream>>>(x, F(PF + 0), F(PF + 1), xn, cHID);
  gemm(xn, cHID, w_fin1, hg,  F(PF + 3), cTOK, cFF,  2);
  gemm(hg, cFF, w_fin2, agg, F(PF + 5), cTOK, cHID, 0);
  final_out_kernel<<<blk(cTOK * cHID), TPB, 0, stream>>>(x, agg, maskq, (float*)d_out);
}
